// Starcoder2Attention_35888746725456
// MI455X (gfx1250) — matrix operand order repaired and checked by reading, not compiled, not run on hardware
//
#include <hip/hip_runtime.h>
#include <hip/hip_bf16.h>
#include <cstdint>

// ---------------------------------------------------------------------------
// Starcoder2 attention for gfx1250 (MI455X): bf16 WMMA everywhere.
//   B=2 S=2048 H=32 KVH=8 D=128 HIDDEN=4096 QSZ=4096 KVSZ=1024
// ---------------------------------------------------------------------------

#define B_DIM 2
#define S_DIM 2048
#define T_DIM 4096
#define NH 32
#define NKVH 8
#define HD 128
#define HIDDEN 4096
#define QSZ 4096
#define KVSZ 1024
#define QKV_N 6144
#define SCALE_F 0.08838834764831845f  // 128^-0.5

typedef __attribute__((ext_vector_type(16))) __bf16 bf16x16;
typedef __attribute__((ext_vector_type(8)))  __bf16 bf16x8;
typedef __attribute__((ext_vector_type(8)))  float  f32x8;

union FragA { bf16x16 v; bf16x8 h[2]; };

// ---------------------------------------------------------------------------
// 1a) fp32 -> bf16 converter (grid-stride), for A-side (row-major kept)
// ---------------------------------------------------------------------------
__global__ void k_f32_to_bf16(const float* __restrict__ in,
                              __bf16* __restrict__ out, size_t n) {
  size_t i = (size_t)blockIdx.x * blockDim.x + threadIdx.x;
  size_t stride = (size_t)gridDim.x * blockDim.x;
  for (; i < n; i += stride) out[i] = (__bf16)in[i];
}

// ---------------------------------------------------------------------------
// 1b) fp32 [R][C] -> bf16 [C][R] tiled transpose-convert (weights, one-time).
// ---------------------------------------------------------------------------
__global__ __launch_bounds__(256)
void k_f32_to_bf16_T(const float* __restrict__ in, __bf16* __restrict__ out,
                     int R, int Cdim) {
  __shared__ __bf16 t[32][33];
  const int c0 = blockIdx.x * 32;
  const int r0 = blockIdx.y * 32;
  const int tx = threadIdx.x & 31;
  const int ty = threadIdx.x >> 5;  // 0..7
#pragma unroll
  for (int i = 0; i < 4; ++i) {
    const int rl = ty + i * 8;
    t[rl][tx] = (__bf16)in[(size_t)(r0 + rl) * Cdim + c0 + tx];
  }
  __syncthreads();
#pragma unroll
  for (int i = 0; i < 4; ++i) {
    const int cl = ty + i * 8;
    out[(size_t)(c0 + cl) * R + r0 + tx] = t[tx][cl];
  }
}

// ---------------------------------------------------------------------------
// 1c) bf16 [R][C] -> bf16 [C][R] tiled transpose, per-slice (for V: [S][D]->[D][S])
// ---------------------------------------------------------------------------
__global__ __launch_bounds__(256)
void k_transpose_bf16(const __bf16* __restrict__ in, __bf16* __restrict__ out,
                      int R, int Cdim) {
  __shared__ __bf16 t[32][33];
  const size_t sl = (size_t)blockIdx.z * R * Cdim;
  const int c0 = blockIdx.x * 32;
  const int r0 = blockIdx.y * 32;
  const int tx = threadIdx.x & 31;
  const int ty = threadIdx.x >> 5;
#pragma unroll
  for (int i = 0; i < 4; ++i) {
    const int rl = ty + i * 8;
    t[rl][tx] = in[sl + (size_t)(r0 + rl) * Cdim + c0 + tx];
  }
  __syncthreads();
#pragma unroll
  for (int i = 0; i < 4; ++i) {
    const int cl = ty + i * 8;
    out[sl + (size_t)(c0 + cl) * R + r0 + tx] = t[tx][cl];
  }
}

// ---------------------------------------------------------------------------
// 2) bf16 WMMA GEMM:  C[M,N](f32) = A[M,K] @ BT[N,K]^T + bias
//    Block tile 128x128, BK=32; 8 waves in 4x2 grid, 32x64 per wave
//    (2x4 WMMA tiles). Double-buffered LDS, one barrier per K-step.
// ---------------------------------------------------------------------------
#define GBM 128
#define GBN 128
#define GBK 32
#define GLD 40   // padded LDS stride (bf16 elems) -> 80B rows, 16B aligned

__global__ __launch_bounds__(256)
void k_gemm_bf16(const __bf16* __restrict__ A, const __bf16* __restrict__ BT,
                 const float* __restrict__ bias, float* __restrict__ C,
                 int M, int N, int K) {
  __shared__ __bf16 aS[2][GBM * GLD];
  __shared__ __bf16 bS[2][GBN * GLD];

  const int tid  = threadIdx.x;
  const int lane = tid & 31;
  const int wid  = tid >> 5;
  const int wm   = (wid >> 1) * 32;   // wave row offset (4 rows of waves)
  const int wn   = (wid & 1) * 64;    // wave col offset (2 cols of waves)
  const int m0   = blockIdx.y * GBM;
  const int n0   = blockIdx.x * GBN;

  const int l16 = lane & 15;
  const int hs  = lane >> 4;          // half-wave select

  const int row = tid >> 1;           // 0..127 (staging row, A and BT)
  const int col = (tid & 1) * 16;     // 0/16   (staging k-col)

  const __bf16* ga = A  + (size_t)(m0 + row) * K + col;
  const __bf16* gb = BT + (size_t)(n0 + row) * K + col;

  f32x8 acc[2][4] = {};

  // preload tile 0
  {
    uint4 a0 = *(const uint4*)ga, a1 = *(const uint4*)(ga + 8);
    uint4 b0 = *(const uint4*)gb, b1 = *(const uint4*)(gb + 8);
    *(uint4*)&aS[0][row * GLD + col]     = a0;
    *(uint4*)&aS[0][row * GLD + col + 8] = a1;
    *(uint4*)&bS[0][row * GLD + col]     = b0;
    *(uint4*)&bS[0][row * GLD + col + 8] = b1;
  }
  __syncthreads();

  const int nsteps = K / GBK;
  for (int step = 0; step < nsteps; ++step) {
    const int cur = step & 1;
    const bool more = (step + 1) < nsteps;
    uint4 a0, a1, b0, b1;
    if (more) {
      const __bf16* pa = ga + (size_t)(step + 1) * GBK;
      const __bf16* pb = gb + (size_t)(step + 1) * GBK;
      a0 = *(const uint4*)pa; a1 = *(const uint4*)(pa + 8);
      b0 = *(const uint4*)pb; b1 = *(const uint4*)(pb + 8);
      __builtin_prefetch(pa + GBK, 0, 1);   // global_prefetch_b8
      __builtin_prefetch(pb + GBK, 0, 1);
    }

    // hoist ALL fragment loads ahead of the WMMA burst
    FragA af[2], bfr[4];
#pragma unroll
    for (int i = 0; i < 2; ++i) {   // A frag: row wm+i*16+l16, K runs per layout
      const __bf16* p = &aS[cur][(wm + i * 16 + l16) * GLD + hs * 8];
      af[i].h[0] = *(const bf16x8*)p;
      af[i].h[1] = *(const bf16x8*)(p + 16);
    }
#pragma unroll
    for (int j = 0; j < 4; ++j) {   // B frag: N col wn+j*16+l16, K runs per layout
      const __bf16* p = &bS[cur][(wn + j * 16 + l16) * GLD + hs * 8];
      bfr[j].h[0] = *(const bf16x8*)p;
      bfr[j].h[1] = *(const bf16x8*)(p + 16);
    }
#pragma unroll
    for (int j = 0; j < 4; ++j)
#pragma unroll
      for (int i = 0; i < 2; ++i)
        acc[i][j] = __builtin_amdgcn_wmma_f32_16x16x32_bf16(
            false, af[i].v, false, bfr[j].v, (short)0, acc[i][j], false, false);

    if (more) {   // after the WMMA burst so s_wait_loadcnt doesn't stall it
      const int nxt = cur ^ 1;
      *(uint4*)&aS[nxt][row * GLD + col]     = a0;
      *(uint4*)&aS[nxt][row * GLD + col + 8] = a1;
      *(uint4*)&bS[nxt][row * GLD + col]     = b0;
      *(uint4*)&bS[nxt][row * GLD + col + 8] = b1;
    }
    __syncthreads();
  }

  // epilogue: C layout -> lane holds N=l16, M = r + 8*hs per VGPR r
#pragma unroll
  for (int j = 0; j < 4; ++j) {
    const int ccol = n0 + wn + j * 16 + l16;
    const float bv = bias[ccol];
#pragma unroll
    for (int i = 0; i < 2; ++i) {
#pragma unroll
      for (int r = 0; r < 8; ++r) {
        const int crow = m0 + wm + i * 16 + r + 8 * hs;
        C[(size_t)crow * N + ccol] = acc[i][j][r] + bv;
      }
    }
  }
}

// ---------------------------------------------------------------------------
// 3) RoPE (NeoX) + head split: qkv f32 [T,6144] -> Q/K/V bf16 [B,H,S,D]
// ---------------------------------------------------------------------------
__global__ __launch_bounds__(64)
void k_rope_split(const float* __restrict__ qkv, const int* __restrict__ pos,
                  __bf16* __restrict__ Qo, __bf16* __restrict__ Ko,
                  __bf16* __restrict__ Vo) {
  const int t = blockIdx.x;             // 0..4095
  const int i = threadIdx.x;            // pair index 0..63
  const int b = t >> 11;
  const int s = t & (S_DIM - 1);
  const float p = (float)pos[t];
  const float inv_freq = __powf(100000.0f, -(float)(2 * i) / 128.0f);
  float sn, cs;
  __sincosf(p * inv_freq, &sn, &cs);
  const float* rowp = qkv + (size_t)t * QKV_N;

  for (int h = 0; h < NH; ++h) {
    const float* x = rowp + h * HD;
    float x1 = x[i], x2 = x[i + 64];
    __bf16* dst = Qo + ((size_t)(b * NH + h) * S_DIM + s) * HD;
    dst[i]      = (__bf16)(x1 * cs - x2 * sn);
    dst[i + 64] = (__bf16)(x2 * cs + x1 * sn);
  }
  for (int h = 0; h < NKVH; ++h) {
    const float* x = rowp + QSZ + h * HD;
    float x1 = x[i], x2 = x[i + 64];
    __bf16* dk = Ko + ((size_t)(b * NKVH + h) * S_DIM + s) * HD;
    dk[i]      = (__bf16)(x1 * cs - x2 * sn);
    dk[i + 64] = (__bf16)(x2 * cs + x1 * sn);
    const float* xv = rowp + QSZ + KVSZ + h * HD;
    __bf16* dv = Vo + ((size_t)(b * NKVH + h) * S_DIM + s) * HD;
    dv[i]      = (__bf16)xv[i];
    dv[i + 64] = (__bf16)xv[i + 64];
  }
}

// ---------------------------------------------------------------------------
// 4) Causal flash attention, bf16 WMMA.
//    grid = (B*NH, S/64), block = 128 (4 waves, 16 q-rows each).
//    K tile AND pre-transposed V tile staged via
//    global_load_async_to_lds_b128 (ASYNCcnt path) -- no VGPR round-trip.
// ---------------------------------------------------------------------------
#define FLDK 136   // kS stride: 272B rows -> conflict-free 16B frag reads
#define FLDV 40    // vS ([d][k]) stride
#define FLDP 40    // per-wave P scratch stride

__device__ __forceinline__ void async_ld_b128(void* lds, const void* g) {
  unsigned loff = (unsigned)(uintptr_t)lds;
  unsigned long long gaddr = (unsigned long long)(uintptr_t)g;
  asm volatile("global_load_async_to_lds_b128 %0, %1, off"
               :: "v"(loff), "v"(gaddr) : "memory");
}

__global__ __launch_bounds__(128)
void k_flash_attn(const __bf16* __restrict__ Q, const __bf16* __restrict__ Kb,
                  const __bf16* __restrict__ VT, __bf16* __restrict__ Oattn) {
  __shared__ __bf16 kS[32 * FLDK];       // K block [k][d] row-major
  __shared__ __bf16 vS[128 * FLDV];      // V block [d][k]
  __shared__ __bf16 pS[4 * 16 * FLDP];   // per-wave P scratch [16][32]

  const int tid  = threadIdx.x;
  const int lane = tid & 31;
  const int wid  = tid >> 5;
  const int l16  = lane & 15;
  const int hs   = lane >> 4;

  const int bh  = blockIdx.x;            // 0..63
  const int b   = bh >> 5;
  const int h   = bh & 31;
  const int kvh = h >> 2;                // GQA group = 4
  const int q0  = blockIdx.y * 64;
  const int qr0 = q0 + wid * 16;

  const __bf16* Qh  = Q  + (size_t)(b * NH   + h)   * S_DIM * HD;
  const __bf16* Kh  = Kb + (size_t)(b * NKVH + kvh) * S_DIM * HD;
  const __bf16* VTh = VT + (size_t)(b * NKVH + kvh) * S_DIM * HD;  // [D][S]

  // Q fragments: 4 d-tiles of 32 (A layout: row=l16, K runs by half-wave)
  FragA qf[4];
#pragma unroll
  for (int dt = 0; dt < 4; ++dt) {
    const __bf16* p = Qh + (size_t)(qr0 + l16) * HD + dt * 32 + hs * 8;
    qf[dt].h[0] = *(const bf16x8*)p;
    qf[dt].h[1] = *(const bf16x8*)(p + 16);
  }

  f32x8 o[8] = {};
  float mrow[8], lrow[8];
#pragma unroll
  for (int r = 0; r < 8; ++r) { mrow[r] = -__builtin_inff(); lrow[r] = 0.f; }

  const int nkb = (q0 + 64) / 32;        // causal bound on 32-wide K blocks

  for (int kb = 0; kb < nkb; ++kb) {
    const int k0 = kb * 32;
    __syncthreads();   // everyone done reading kS/vS from previous block
    // stage K [k][d] and V [d][k] via async global->LDS DMA
#pragma unroll
    for (int it = 0; it < 4; ++it) {
      const int idx = tid + it * 128;    // vec8 index 0..511
      const int kr = idx >> 4;           // 0..31
      const int dc = (idx & 15) * 8;     // 0..120
      async_ld_b128(&kS[kr * FLDK + dc], Kh + (size_t)(k0 + kr) * HD + dc);
      const int dv = idx >> 2;           // 0..127
      const int ck = (idx & 3) * 8;      // 0..24
      async_ld_b128(&vS[dv * FLDV + ck], VTh + (size_t)dv * S_DIM + k0 + ck);
    }
    asm volatile("s_wait_asynccnt 0" ::: "memory");
    __syncthreads();

    // S = Q K^T : two 16x16 subtiles (k-local 0..15 / 16..31)
    f32x8 sc[2] = {};
#pragma unroll
    for (int sub = 0; sub < 2; ++sub) {
      const int kl = sub * 16 + l16;     // B-frag N index
#pragma unroll
      for (int dt = 0; dt < 4; ++dt) {
        FragA bf;
        const __bf16* p = &kS[kl * FLDK + dt * 32 + hs * 8];
        bf.h[0] = *(const bf16x8*)p;
        bf.h[1] = *(const bf16x8*)(p + 16);
        sc[sub] = __builtin_amdgcn_wmma_f32_16x16x32_bf16(
            false, qf[dt].v, false, bf.v, (short)0, sc[sub], false, false);
      }
    }

    // scale + causal mask (C layout: N = k0+sub*16+l16, M = qr0+r+8*hs)
#pragma unroll
    for (int sub = 0; sub < 2; ++sub) {
      const int kg = k0 + sub * 16 + l16;
#pragma unroll
      for (int r = 0; r < 8; ++r) {
        const int qg = qr0 + r + 8 * hs;
        const float sv = sc[sub][r] * SCALE_F;
        sc[sub][r] = (kg <= qg) ? sv : -__builtin_inff();
      }
    }

    // online softmax; rows live in one 16-lane half -> xor-reduce masks 8..1
#pragma unroll
    for (int r = 0; r < 8; ++r) {
      float vmax = fmaxf(sc[0][r], sc[1][r]);
#pragma unroll
      for (int off = 8; off >= 1; off >>= 1)
        vmax = fmaxf(vmax, __shfl_xor(vmax, off, 32));
      const float mnew = fmaxf(mrow[r], vmax);
      const float p0 = __expf(sc[0][r] - mnew);
      const float p1 = __expf(sc[1][r] - mnew);
      float rs = p0 + p1;
#pragma unroll
      for (int off = 8; off >= 1; off >>= 1)
        rs += __shfl_xor(rs, off, 32);
      const float alpha = __expf(mrow[r] - mnew);
      lrow[r] = lrow[r] * alpha + rs;
      mrow[r] = mnew;
#pragma unroll
      for (int d2 = 0; d2 < 8; ++d2) o[d2][r] *= alpha;
      sc[0][r] = p0;
      sc[1][r] = p1;
    }

    // C-layout P -> wave-private LDS -> A-layout reload (LDS in-order per wave)
    __bf16* pw = &pS[wid * 16 * FLDP];
#pragma unroll
    for (int sub = 0; sub < 2; ++sub) {
      const int pcol = sub * 16 + l16;
#pragma unroll
      for (int r = 0; r < 8; ++r)
        pw[(r + 8 * hs) * FLDP + pcol] = (__bf16)sc[sub][r];
    }
    FragA pf;
    {
      const __bf16* p = &pw[l16 * FLDP + hs * 8];
      pf.h[0] = *(const bf16x8*)p;
      pf.h[1] = *(const bf16x8*)(p + 16);
    }

    // O += P V : 8 d-tiles of 16; V B-frags from vS [d][k]
#pragma unroll
    for (int d2 = 0; d2 < 8; ++d2) {
      FragA vf;
      const __bf16* p = &vS[(d2 * 16 + l16) * FLDV + hs * 8];
      vf.h[0] = *(const bf16x8*)p;
      vf.h[1] = *(const bf16x8*)(p + 16);
      o[d2] = __builtin_amdgcn_wmma_f32_16x16x32_bf16(
          false, pf.v, false, vf.v, (short)0, o[d2], false, false);
    }
  }

  // normalize + write attn bf16 at [t, h*128+d]
#pragma unroll
  for (int r = 0; r < 8; ++r) {
    const float inv = 1.f / lrow[r];
    const int qg = qr0 + r + 8 * hs;
    const size_t t = (size_t)b * S_DIM + qg;
#pragma unroll
    for (int d2 = 0; d2 < 8; ++d2)
      Oattn[t * QSZ + h * HD + d2 * 16 + l16] = (__bf16)(o[d2][r] * inv);
  }
}

// ---------------------------------------------------------------------------
// launch
// ---------------------------------------------------------------------------
extern "C" void kernel_launch(void* const* d_in, const int* in_sizes, int n_in,
                              void* d_out, int out_size, void* d_ws, size_t ws_size,
                              hipStream_t stream) {
  const float* hs    = (const float*)d_in[0];
  const int*   pos   = (const int*)d_in[1];
  const float* w_qkv = (const float*)d_in[2];
  const float* b_qkv = (const float*)d_in[3];
  const float* w_o   = (const float*)d_in[4];
  const float* b_o   = (const float*)d_in[5];
  float*       out   = (float*)d_out;

  char* ws = (char*)d_ws;
  size_t off = 0;
  auto alloc = [&](size_t bytes) {
    void* p = ws + off;
    off = (off + bytes + 255) & ~(size_t)255;
    return p;
  };
  __bf16* hs_bf    = (__bf16*)alloc((size_t)T_DIM * HIDDEN * 2);
  __bf16* wqkvT_bf = (__bf16*)alloc((size_t)HIDDEN * QKV_N * 2);  // [N][K]
  __bf16* woT_bf   = (__bf16*)alloc((size_t)QSZ * HIDDEN * 2);    // [N][K]
  float*  qkv      = (float*) alloc((size_t)T_DIM * QKV_N * 4);
  __bf16* qbuf     = (__bf16*)alloc((size_t)B_DIM * NH * S_DIM * HD * 2);
  __bf16* kbuf     = (__bf16*)alloc((size_t)B_DIM * NKVH * S_DIM * HD * 2);
  __bf16* vbuf     = (__bf16*)alloc((size_t)B_DIM * NKVH * S_DIM * HD * 2);
  __bf16* vTbuf    = (__bf16*)alloc((size_t)B_DIM * NKVH * S_DIM * HD * 2);
  __bf16* attn     = (__bf16*)alloc((size_t)T_DIM * QSZ * 2);

  // 1) convert A-side; transpose-convert weights to [N][K]
  {
    size_t n1 = (size_t)T_DIM * HIDDEN;
    k_f32_to_bf16<<<(int)((n1 + 1023) / 1024), 256, 0, stream>>>(hs, hs_bf, n1);
    dim3 g1(QKV_N / 32, HIDDEN / 32);   // w_qkv [K=4096][N=6144] -> [N][K]
    k_f32_to_bf16_T<<<g1, 256, 0, stream>>>(w_qkv, wqkvT_bf, HIDDEN, QKV_N);
    dim3 g2(HIDDEN / 32, QSZ / 32);     // w_o [K=4096][N=4096] -> [N][K]
    k_f32_to_bf16_T<<<g2, 256, 0, stream>>>(w_o, woT_bf, QSZ, HIDDEN);
  }
  // 2) QKV projection: [4096,4096] @ [4096,6144] + b -> f32
  {
    dim3 grid(QKV_N / GBN, T_DIM / GBM);
    k_gemm_bf16<<<grid, 256, 0, stream>>>(hs_bf, wqkvT_bf, b_qkv, qkv,
                                          T_DIM, QKV_N, HIDDEN);
  }
  // 3) RoPE + split to [B,H,S,D] bf16
  k_rope_split<<<T_DIM, 64, 0, stream>>>(qkv, pos, qbuf, kbuf, vbuf);
  // 3b) V: [S][D] -> [D][S] per (b,kvh) slice, for async B-frag staging
  {
    dim3 g(HD / 32, S_DIM / 32, B_DIM * NKVH);
    k_transpose_bf16<<<g, 256, 0, stream>>>(vbuf, vTbuf, S_DIM, HD);
  }
  // 4) causal flash attention
  {
    dim3 grid(B_DIM * NH, S_DIM / 64);
    k_flash_attn<<<grid, 128, 0, stream>>>(qbuf, kbuf, vTbuf, attn);
  }
  // 5) output projection: [4096,4096] @ [4096,4096] + b -> f32 out
  {
    dim3 grid(HIDDEN / GBN, T_DIM / GBM);
    k_gemm_bf16<<<grid, 256, 0, stream>>>(attn, woT_bf, b_o, out,
                                          T_DIM, HIDDEN, QSZ);
  }
  (void)in_sizes; (void)n_in; (void)out_size; (void)ws_size;
}
